// GNN_node_28432683499897
// MI455X (gfx1250) — matrix operand order repaired and checked by reading, not compiled, not run on hardware
//
#include <hip/hip_runtime.h>
#include <hip/hip_bf16.h>

typedef __attribute__((ext_vector_type(16))) __bf16 bf16x16;
typedef __attribute__((ext_vector_type(2)))  __bf16 bf16x2;
typedef __attribute__((ext_vector_type(8)))  float  f32x8;

#define D_FEAT 128
#define D_HID  256

// ---------------------------------------------------------------- helpers ---
__device__ inline float4 ld4(const float* __restrict__ p) {
    return *(const float4*)p;
}

__device__ inline void atomAddF(float* p, float v) {
    __hip_atomic_fetch_add(p, v, __ATOMIC_RELAXED, __HIP_MEMORY_SCOPE_AGENT);
}

// fused BN-affine + ReLU on a float4 (used for GEMM2's A operand)
__device__ inline float4 bn_relu4(float4 v, float4 s, float4 t) {
    float4 r;
    r.x = fmaxf(fmaf(v.x, s.x, t.x), 0.f);
    r.y = fmaxf(fmaf(v.y, s.y, t.y), 0.f);
    r.z = fmaxf(fmaf(v.z, s.z, t.z), 0.f);
    r.w = fmaxf(fmaf(v.w, s.w, t.w), 0.f);
    return r;
}

// ------------------------------------------------------------ init h ---------
// h[n,d] = sum_f atom_emb[f, x[n,f], d]
__global__ void init_h_kernel(const int* __restrict__ x,
                              const float* __restrict__ atom_emb,
                              float* __restrict__ h, int N)
{
    int idx = blockIdx.x * blockDim.x + threadIdx.x;
    int n = idx >> 7;
    int d = idx & 127;
    if (n >= N) return;
    const int* xr = x + n * 9;
    float s = 0.f;
#pragma unroll
    for (int f = 0; f < 9; ++f) {
        int v = xr[f];
        s += atom_emb[(f * 64 + v) * D_FEAT + d];
    }
    h[n * D_FEAT + d] = s;
}

// z = (1 + eps_p[l]) * h     (float4 per thread)
__global__ void scale_h_kernel(const float* __restrict__ h,
                               const float* __restrict__ eps_p, int l,
                               float* __restrict__ z, int total4)
{
    int idx = blockIdx.x * blockDim.x + threadIdx.x;
    if (idx >= total4) return;
    float s = 1.0f + eps_p[l];
    float4 v = ld4(h + (size_t)idx * 4);
    v.x *= s; v.y *= s; v.z *= s; v.w *= s;
    *(float4*)(z + (size_t)idx * 4) = v;
}

// ----------------------------------------------------- edge message+scatter --
// 32 lanes per edge, float4 per lane.
// msg = relu(h[src] + sum_f bond[f, attr_f]) * ew ; z[dst] += msg (atomic)
__global__ void edge_message_kernel(const float* __restrict__ h,
                                    const int* __restrict__ edge_index,
                                    const int* __restrict__ edge_attr,
                                    const float* __restrict__ ew,
                                    const float* __restrict__ bond, // 3*8*128 this layer
                                    float* __restrict__ z, int E)
{
    int e  = blockIdx.x * (blockDim.x >> 5) + (threadIdx.x >> 5);
    int d4 = (threadIdx.x & 31) * 4;
    if (e >= E) return;
    int src = edge_index[e];
    int dst = edge_index[E + e];
    int a0 = edge_attr[e * 3 + 0];
    int a1 = edge_attr[e * 3 + 1];
    int a2 = edge_attr[e * 3 + 2];
    float w = ew[e];
    float4 b0 = ld4(bond + (0 * 8 + a0) * D_FEAT + d4);
    float4 b1 = ld4(bond + (1 * 8 + a1) * D_FEAT + d4);
    float4 b2 = ld4(bond + (2 * 8 + a2) * D_FEAT + d4);
    float4 hv = ld4(h + (size_t)src * D_FEAT + d4);
    float4 m;
    m.x = fmaxf(hv.x + b0.x + b1.x + b2.x, 0.f) * w;
    m.y = fmaxf(hv.y + b0.y + b1.y + b2.y, 0.f) * w;
    m.z = fmaxf(hv.z + b0.z + b1.z + b2.z, 0.f) * w;
    m.w = fmaxf(hv.w + b0.w + b1.w + b2.w, 0.f) * w;
    float* zp = z + (size_t)dst * D_FEAT + d4;
    atomAddF(zp + 0, m.x);
    atomAddF(zp + 1, m.y);
    atomAddF(zp + 2, m.z);
    atomAddF(zp + 3, m.w);
}

// ------------------------------------------------- weight fragment packing ---
// Pack W (K x Ncol, f32 row-major) into bf16 WMMA B-fragments so each lane's
// 16 elements are contiguous (32B vector load in the GEMM):
//   out[((nt*kSteps + kb)*32 + lane)*16 + e] = bf16( W[k*Ncol + n] )
//   kh=lane>>4, nc=lane&15, n=nt*16+nc, k=kb*32+kh*16+e
__global__ void pack_w_kernel(const float* __restrict__ W,
                              __bf16* __restrict__ out, int K, int Ncol)
{
    int idx = blockIdx.x * blockDim.x + threadIdx.x;
    if (idx >= K * Ncol) return;
    int e    = idx & 15;
    int lane = (idx >> 4) & 31;
    int frag = idx >> 9;
    int kSteps = K >> 5;
    int nt = frag / kSteps;
    int kb = frag - nt * kSteps;
    int kh = lane >> 4, nc = lane & 15;
    int n = nt * 16 + nc;
    int k = kb * 32 + kh * 16 + e;
    out[idx] = (__bf16)W[(size_t)k * Ncol + n];
}

// ------------------------------------------------------------- WMMA GEMM -----
// C[M,NCOL] = op(A[M,K]) @ W[K,NCOL] ; W pre-packed bf16 fragments.
// op(A) = identity if aScale==nullptr, else relu(A*scale[k]+shift[k]).
// Block = 128 threads (4 waves). The block's 16-row A tile is staged ONCE into
// LDS -- transformed, bf16-converted, and swizzled into WMMA A-fragment layout
// -- then every wave fetches each A fragment as one contiguous 32B ds_load.
// Each wave owns NT/4 output tiles; K loop fully unrolled.
template<int K, int NCOL>
__global__ void __launch_bounds__(128)
wmma_gemm_kernel(const float* __restrict__ A, const __bf16* __restrict__ Wp,
                 float* __restrict__ C, int M,
                 const float* __restrict__ aScale,
                 const float* __restrict__ aShift)
{
    constexpr int KSTEPS = K / 32;
    constexpr int NT     = NCOL / 16;
    constexpr int TPW    = NT / 4;          // output tiles per wave
    __shared__ __align__(32) __bf16 aSh[KSTEPS * 32 * 16];   // 16 rows x K, frag layout

    const int tid = threadIdx.x;
    const int m0  = blockIdx.x * 16;

    // ---- stage A tile (16 x K) into LDS in WMMA fragment layout ------------
    // thread covers row = tid/8, K/8 consecutive cols starting (tid%8)*(K/8)
    {
        const int row = tid >> 3;
        const int c0  = (tid & 7) * (K / 8);
        int m = m0 + row; if (m >= M) m = M - 1;
        const float* Ar = A + (size_t)m * K;
#pragma unroll
        for (int c = 0; c < K / 8; c += 4) {
            int k = c0 + c;
            float4 v = ld4(Ar + k);
            if (aScale)     // fused BN1-apply + ReLU, done once per element
                v = bn_relu4(v, ld4(aScale + k), ld4(aShift + k));
            // (row, k..k+3) -> fragment [kb][lane][e] ; e,e+1,e+2,e+3 contiguous
            int kb   = k >> 5;
            int r32  = k & 31;
            int kh   = r32 >> 4;
            int e    = r32 & 15;
            int lane = kh * 16 + row;
            __bf16* dst = &aSh[(kb * 32 + lane) * 16 + e];
            bf16x2 p0; p0[0] = (__bf16)v.x; p0[1] = (__bf16)v.y;
            bf16x2 p1; p1[0] = (__bf16)v.z; p1[1] = (__bf16)v.w;
            *(bf16x2*)(dst)     = p0;
            *(bf16x2*)(dst + 2) = p1;
        }
    }
    __syncthreads();

    const int lane   = tid & 31;
    const int wave   = tid >> 5;
    const int kh     = lane >> 4;
    const int lr     = lane & 15;
    const int ntBase = wave * TPW;

    f32x8 acc[TPW] = {};
#pragma unroll
    for (int kb = 0; kb < KSTEPS; ++kb) {
        bf16x16 afrag = *(const bf16x16*)&aSh[(kb * 32 + lane) * 16]; // 32B ds_load
#pragma unroll
        for (int t = 0; t < TPW; ++t) {
            int nt = ntBase + t;
            const __bf16* Bp = Wp + (((size_t)nt * KSTEPS + kb) * 32 + lane) * 16;
            bf16x16 bfrag = *(const bf16x16*)Bp;   // 32B contiguous, coalesced
            acc[t] = __builtin_amdgcn_wmma_f32_16x16x32_bf16(
                         false, afrag, false, bfrag, (short)0, acc[t],
                         false, false);
        }
    }

    // ---- D store: VGPR j -> row m0 + j + kh*8, column n ----
    int mbase = m0 + kh * 8;
#pragma unroll
    for (int t = 0; t < TPW; ++t) {
        int n = (ntBase + t) * 16 + lr;
#pragma unroll
        for (int j = 0; j < 8; ++j) {
            int mm = mbase + j;
            if (mm < M) C[(size_t)mm * NCOL + n] = acc[t][j];
        }
    }
}

// -------------------------------------------------------------- BN pieces ----
__global__ void zero_kernel(float* __restrict__ p, int n)
{
    int i = blockIdx.x * blockDim.x + threadIdx.x;
    if (i < n) p[i] = 0.f;
}

// per-column sum / sumsq ; blockDim.x == Cc ; coalesced row-major sweep
__global__ void col_stats_kernel(const float* __restrict__ t, int M, int Cc,
                                 float* __restrict__ sums)
{
    int j = threadIdx.x;
    float s = 0.f, s2 = 0.f;
    for (int r = blockIdx.x; r < M; r += gridDim.x) {
        __builtin_prefetch(t + (size_t)(r + gridDim.x) * Cc + j, 0, 1);
        float v = t[(size_t)r * Cc + j];
        s += v; s2 += v * v;
    }
    atomAddF(&sums[j], s);
    atomAddF(&sums[Cc + j], s2);
}

// fold mean/var/gamma/beta into per-column affine: y = x*scale + shift
// (linear-layer biases cancel inside BN, so they are never added)
__global__ void bn_finalize_kernel(const float* __restrict__ sums,
                                   const float* __restrict__ g,
                                   const float* __restrict__ b,
                                   int M, int Cc,
                                   float* __restrict__ scale,
                                   float* __restrict__ shift)
{
    int j = threadIdx.x;
    if (j >= Cc) return;
    float inv  = 1.0f / (float)M;
    float mean = sums[j] * inv;
    float var  = sums[Cc + j] * inv - mean * mean;
    float sc   = g[j] * rsqrtf(var + 1e-5f);
    scale[j] = sc;
    shift[j] = b[j] - mean * sc;
}

// h = maybe_relu(z2*scale + shift)   (float4 per thread; D=128 -> 32 vec4/row)
__global__ void bn_apply_kernel(const float* __restrict__ zin,
                                const float* __restrict__ scale,
                                const float* __restrict__ shift,
                                int doRelu, float* __restrict__ hout, int total4)
{
    int idx = blockIdx.x * blockDim.x + threadIdx.x;
    if (idx >= total4) return;
    int d4 = (idx & 31) * 4;
    float4 v  = ld4(zin + (size_t)idx * 4);
    float4 sc = ld4(scale + d4);
    float4 sh = ld4(shift + d4);
    v.x = fmaf(v.x, sc.x, sh.x);
    v.y = fmaf(v.y, sc.y, sh.y);
    v.z = fmaf(v.z, sc.z, sh.z);
    v.w = fmaf(v.w, sc.w, sh.w);
    if (doRelu) {
        v.x = fmaxf(v.x, 0.f); v.y = fmaxf(v.y, 0.f);
        v.z = fmaxf(v.z, 0.f); v.w = fmaxf(v.w, 0.f);
    }
    *(float4*)(hout + (size_t)idx * 4) = v;
}

// ---------------------------------------------------------------- launch -----
extern "C" void kernel_launch(void* const* d_in, const int* in_sizes, int n_in,
                              void* d_out, int out_size, void* d_ws, size_t ws_size,
                              hipStream_t stream)
{
    (void)n_in; (void)out_size; (void)ws_size;
    const int*   x          = (const int*)  d_in[0];
    const int*   edge_index = (const int*)  d_in[1];
    const int*   edge_attr  = (const int*)  d_in[2];
    const float* edge_w     = (const float*)d_in[3];
    const float* atom_emb   = (const float*)d_in[4];
    const float* bond_emb   = (const float*)d_in[5];
    const float* W1         = (const float*)d_in[6];
    /* b1 = d_in[7]  : cancels inside BN1 */
    const float* bn1_g      = (const float*)d_in[8];
    const float* bn1_b      = (const float*)d_in[9];
    const float* W2         = (const float*)d_in[10];
    /* b2 = d_in[11] : cancels inside BN2 */
    const float* eps_p      = (const float*)d_in[12];
    const float* bn_g       = (const float*)d_in[13];
    const float* bn_b       = (const float*)d_in[14];

    const int N = in_sizes[0] / 9;
    const int E = in_sizes[3];
    const int L = in_sizes[12];

    float* h     = (float*)d_out;                  // N x 128 (ping buffer)
    float* z     = (float*)d_ws;                   // N x 128
    float* t     = z  + (size_t)N * D_FEAT;        // N x 256
    float* z2    = t  + (size_t)N * D_HID;         // N x 128
    float* sums  = z2 + (size_t)N * D_FEAT;        // 512
    float* scl   = sums + 2 * D_HID;               // 256
    float* shf   = scl  + D_HID;                   // 256
    __bf16* W1p  = (__bf16*)(shf + D_HID);         // L * 128*256 bf16
    __bf16* W2p  = W1p + (size_t)L * D_FEAT * D_HID;

    const int totD   = N * D_FEAT;
    const int totD4  = totD / 4;
    const dim3 b256(256);
    const int mTiles = (N + 15) / 16;
    const int WELTS  = D_FEAT * D_HID;             // 32768 per layer per matrix

    // Pack all layer weights into bf16 WMMA fragments (once per launch).
    for (int l = 0; l < L; ++l) {
        pack_w_kernel<<<dim3((WELTS + 255) / 256), b256, 0, stream>>>(
            W1 + (size_t)l * WELTS, W1p + (size_t)l * WELTS, D_FEAT, D_HID);
        pack_w_kernel<<<dim3((WELTS + 255) / 256), b256, 0, stream>>>(
            W2 + (size_t)l * WELTS, W2p + (size_t)l * WELTS, D_HID, D_FEAT);
    }

    init_h_kernel<<<dim3((totD + 255) / 256), b256, 0, stream>>>(x, atom_emb, h, N);

    for (int l = 0; l < L; ++l) {
        scale_h_kernel<<<dim3((totD4 + 255) / 256), b256, 0, stream>>>(
            h, eps_p, l, z, totD4);
        edge_message_kernel<<<dim3((E + 7) / 8), b256, 0, stream>>>(
            h, edge_index, edge_attr, edge_w,
            bond_emb + (size_t)l * 3 * 8 * D_FEAT, z, E);

        // GEMM1: t = z @ W1[l]   (128 -> 256); LDS-staged A, 4 tiles/wave
        wmma_gemm_kernel<D_FEAT, D_HID><<<dim3(mTiles), dim3(128), 0, stream>>>(
            z, W1p + (size_t)l * WELTS, t, N, nullptr, nullptr);

        zero_kernel<<<1, 2 * D_HID, 0, stream>>>(sums, 2 * D_HID);
        col_stats_kernel<<<dim3(256), dim3(D_HID), 0, stream>>>(t, N, D_HID, sums);
        bn_finalize_kernel<<<1, D_HID, 0, stream>>>(
            sums, bn1_g + (size_t)l * D_HID, bn1_b + (size_t)l * D_HID,
            N, D_HID, scl, shf);

        // GEMM2: z2 = relu(BN1(t)) @ W2[l]  (BN1+ReLU fused into A staging)
        wmma_gemm_kernel<D_HID, D_FEAT><<<dim3(mTiles), dim3(128), 0, stream>>>(
            t, W2p + (size_t)l * WELTS, z2, N, scl, shf);

        zero_kernel<<<1, 2 * D_HID, 0, stream>>>(sums, 2 * D_HID);
        col_stats_kernel<<<dim3(256), dim3(D_FEAT), 0, stream>>>(z2, N, D_FEAT, sums);
        bn_finalize_kernel<<<1, D_FEAT, 0, stream>>>(
            sums, bn_g + (size_t)l * D_FEAT, bn_b + (size_t)l * D_FEAT,
            N, D_FEAT, scl, shf);

        bn_apply_kernel<<<dim3((totD4 + 255) / 256), b256, 0, stream>>>(
            z2, scl, shf, (l < L - 1) ? 1 : 0, h, totD4);
    }
}
